// LinearAttention_58884001628511
// MI455X (gfx1250) — compile-verified, matrix operand order checked by reading, not run on hardware
//
#include <hip/hip_runtime.h>
#include <cstdint>
#include <cstddef>

#define B_ 4
#define L_ 4096
#define C_ 1024
#define H_ 16
#define D_ 64
#define M_ (B_ * L_) /* 16384 rows */

typedef __attribute__((ext_vector_type(8)))  unsigned short ushort8;
typedef __attribute__((ext_vector_type(16))) __bf16         v16bf;
typedef __attribute__((ext_vector_type(8)))  float          v8f;

union FragBf { ushort8 u[2]; v16bf bf; };

__device__ __forceinline__ unsigned short f2bf(float f) {
  unsigned int u = __float_as_uint(f);
  u += 0x7FFFu + ((u >> 16) & 1u);           // round-to-nearest-even
  return (unsigned short)(u >> 16);
}
__device__ __forceinline__ float bf2f(unsigned short u) {
  return __uint_as_float(((unsigned int)u) << 16);
}

// ---------------------------------------------------------------------------
// 1) f32 -> bf16 elementwise convert
// ---------------------------------------------------------------------------
__global__ __launch_bounds__(256) void cvt_bf16_kernel(
    const float* __restrict__ in, unsigned short* __restrict__ out, size_t n) {
  size_t i = (size_t)blockIdx.x * blockDim.x + threadIdx.x;
  if (i < n) out[i] = f2bf(in[i]);
}

// ---------------------------------------------------------------------------
// 2) weight transpose-convert: Wt[n*C + k] = bf16(W[k*C + n]); LDS 32x32 tile
// ---------------------------------------------------------------------------
__global__ __launch_bounds__(256) void transpose_w_kernel(
    const float* __restrict__ W, unsigned short* __restrict__ Wt) {
  __shared__ float tile[32][33];
  const int bx = blockIdx.x * 32;  // n tile
  const int by = blockIdx.y * 32;  // k tile
  const int tx = threadIdx.x;      // 0..31
  const int ty = threadIdx.y;      // 0..7
#pragma unroll
  for (int s = 0; s < 4; ++s)
    tile[ty + 8 * s][tx] = W[(size_t)(by + ty + 8 * s) * C_ + (bx + tx)];
  __syncthreads();
#pragma unroll
  for (int s = 0; s < 4; ++s)
    Wt[(size_t)(bx + ty + 8 * s) * C_ + (by + tx)] = f2bf(tile[tx][ty + 8 * s]);
}

// ---------------------------------------------------------------------------
// 3) GEMM: Cout[M x N] = A[M x K] (bf16, row-major) * Bt[N x K]^T (bf16) + bias
//    block = 256 thr (8 waves); block tile 256(M) x 64(N); wave tile 32 x 64.
//    WMMA bf16 fragments: per-lane row m = lane&15, khalf = (lane>>4)*8,
//    components 0..7 -> K khalf..khalf+7, components 8..15 -> K khalf+16..+23.
// ---------------------------------------------------------------------------
__global__ __launch_bounds__(256) void gemm_bf16_tn_kernel(
    const unsigned short* __restrict__ A, const unsigned short* __restrict__ Bt,
    const float* __restrict__ bias, float* __restrict__ Cout,
    int M, int N, int K) {
  const int lane  = threadIdx.x & 31;
  const int wave  = threadIdx.x >> 5;
  const int m0    = blockIdx.x * 256 + wave * 32;
  const int n0    = blockIdx.y * 64;
  const int lr    = lane & 15;
  const int khalf = (lane >> 4) * 8;

  v8f acc[2][4];
#pragma unroll
  for (int i = 0; i < 2; ++i)
#pragma unroll
    for (int j = 0; j < 4; ++j) acc[i][j] = {};

  for (int k0 = 0; k0 < K; k0 += 32) {
    FragBf aF[2], bF[4];
#pragma unroll
    for (int i = 0; i < 2; ++i) {
      const unsigned short* p = A + (size_t)(m0 + i * 16 + lr) * K + k0 + khalf;
      aF[i].u[0] = *(const ushort8*)(p);
      aF[i].u[1] = *(const ushort8*)(p + 16);
    }
#pragma unroll
    for (int j = 0; j < 4; ++j) {
      const unsigned short* p = Bt + (size_t)(n0 + j * 16 + lr) * K + k0 + khalf;
      bF[j].u[0] = *(const ushort8*)(p);
      bF[j].u[1] = *(const ushort8*)(p + 16);
    }
    if (k0 + 32 < K) {  // global_prefetch_b8 of next K-step
      __builtin_prefetch(A + (size_t)(m0 + lr) * K + k0 + 32, 0, 1);
      __builtin_prefetch(Bt + (size_t)(n0 + lr) * K + k0 + 32, 0, 1);
    }
#pragma unroll
    for (int i = 0; i < 2; ++i)
#pragma unroll
      for (int j = 0; j < 4; ++j)
        acc[i][j] = __builtin_amdgcn_wmma_f32_16x16x32_bf16(
            false, aF[i].bf, false, bF[j].bf, (short)0, acc[i][j], false, false);
  }

  const int rbase = (lane >> 4) * 8;  // D layout: row = r + rbase, col = lr
#pragma unroll
  for (int i = 0; i < 2; ++i)
#pragma unroll
    for (int j = 0; j < 4; ++j) {
      const int n  = n0 + j * 16 + lr;
      const float bv = bias[n];
#pragma unroll
      for (int r = 0; r < 8; ++r) {
        const int m = m0 + i * 16 + rbase + r;
        Cout[(size_t)m * N + n] = acc[i][j][r] + bv;
      }
    }
}

// ---------------------------------------------------------------------------
// 4) softmax over heads (axis=1 of (B,H,L,D)) + pack:
//    qp  bf16, (B*L, C) row-major            (A of the out_num GEMM)
//    kpT bf16, (B,H,D,L) transposed          (A of the kv GEMM, K=l contiguous)
//    vT  bf16, (B,H,D,L) transposed          (B of the kv GEMM)
// ---------------------------------------------------------------------------
__global__ __launch_bounds__(256) void softmax_pack_kernel(
    const float* __restrict__ qf, const float* __restrict__ kf,
    const float* __restrict__ vf, unsigned short* __restrict__ qp,
    unsigned short* __restrict__ kpT, unsigned short* __restrict__ vT) {
  const int idx = blockIdx.x * blockDim.x + threadIdx.x;  // over B*L*D
  if (idx >= B_ * L_ * D_) return;
  const int d  = idx & (D_ - 1);
  const int bl = idx >> 6;
  const int b  = bl / L_;
  const int l  = bl - b * L_;

  float qv[H_], kv[H_];
  float qmax = -3.0e38f, kmax = -3.0e38f;
#pragma unroll
  for (int h = 0; h < H_; ++h) {
    qv[h] = qf[(size_t)bl * C_ + h * D_ + d];
    kv[h] = kf[(size_t)bl * C_ + h * D_ + d];
    qmax = fmaxf(qmax, qv[h]);
    kmax = fmaxf(kmax, kv[h]);
  }
  float qs = 0.f, ks = 0.f;
#pragma unroll
  for (int h = 0; h < H_; ++h) {
    qv[h] = __expf(qv[h] - qmax); qs += qv[h];
    kv[h] = __expf(kv[h] - kmax); ks += kv[h];
  }
  const float qinv = 1.f / qs, kinv = 1.f / ks;
#pragma unroll
  for (int h = 0; h < H_; ++h) {
    qp[(size_t)bl * C_ + h * D_ + d] = f2bf(qv[h] * qinv);
    const size_t trow = ((size_t)(b * H_ + h) * D_ + d) * L_ + l;
    kpT[trow] = f2bf(kv[h] * kinv);
    vT[trow]  = f2bf(vf[(size_t)bl * C_ + h * D_ + d]);
  }
}

// ---------------------------------------------------------------------------
// 5) kv state: kv[b,h,d,e] = sum_l k'[l,d] * v[l,e]; one block per (b,h).
//    A = kpT rows (K=l contiguous), B = vT rows. 16 tiles of 16x16, 2/wave.
//    Stores kvT bf16 e-major (so next GEMM's B-fragments are contiguous).
//    Also reduces ksum[b,h,d] = sum_l k'.
// ---------------------------------------------------------------------------
__global__ __launch_bounds__(256) void kv_state_kernel(
    const unsigned short* __restrict__ kpT, const unsigned short* __restrict__ vT,
    unsigned short* __restrict__ kvT, float* __restrict__ ksum) {
  const int bh   = blockIdx.x;            // 0..63
  const int lane = threadIdx.x & 31;
  const int wave = threadIdx.x >> 5;

  if (threadIdx.x < D_) {                 // k'-sum over l (contiguous row)
    const unsigned short* row = kpT + ((size_t)bh * D_ + threadIdx.x) * L_;
    float s = 0.f;
    for (int l = 0; l < L_; ++l) s += bf2f(row[l]);
    ksum[bh * D_ + threadIdx.x] = s;
  }

  const int lr    = lane & 15;
  const int khalf = (lane >> 4) * 8;
  const int rbase = (lane >> 4) * 8;
#pragma unroll
  for (int t = wave * 2; t < wave * 2 + 2; ++t) {
    const int ti = t >> 2, tj = t & 3;    // d-tile, e-tile
    v8f acc = {};
    const unsigned short* arow = kpT + ((size_t)bh * D_ + ti * 16 + lr) * L_ + khalf;
    const unsigned short* brow = vT  + ((size_t)bh * D_ + tj * 16 + lr) * L_ + khalf;
    for (int l0 = 0; l0 < L_; l0 += 32) {
      FragBf aF, bF;
      aF.u[0] = *(const ushort8*)(arow + l0);
      aF.u[1] = *(const ushort8*)(arow + l0 + 16);
      bF.u[0] = *(const ushort8*)(brow + l0);
      bF.u[1] = *(const ushort8*)(brow + l0 + 16);
      if (l0 + 32 < L_) {
        __builtin_prefetch(arow + l0 + 32, 0, 1);
        __builtin_prefetch(brow + l0 + 32, 0, 1);
      }
      acc = __builtin_amdgcn_wmma_f32_16x16x32_bf16(
          false, aF.bf, false, bF.bf, (short)0, acc, false, false);
    }
#pragma unroll
    for (int r = 0; r < 8; ++r) {
      const int d = ti * 16 + rbase + r;
      const int e = tj * 16 + lr;
      kvT[((size_t)bh * D_ + e) * D_ + d] = f2bf(acc[r]);
    }
  }
}

// ---------------------------------------------------------------------------
// 6) denom[b,h,l] = dot(q'[b,h,l,:], ksum[b,h,:]) + 1e-6
// ---------------------------------------------------------------------------
__global__ __launch_bounds__(256) void denom_kernel(
    const unsigned short* __restrict__ qp, const float* __restrict__ ksum,
    float* __restrict__ denom) {
  const int idx = blockIdx.x * blockDim.x + threadIdx.x;  // bh*L + l
  if (idx >= B_ * H_ * L_) return;
  const int bh = idx / L_;
  const int l  = idx - bh * L_;
  const int b  = bh >> 4, h = bh & 15;
  const unsigned short* qrow = qp + ((size_t)(b * L_ + l)) * C_ + h * D_;
  const float* ks = ksum + bh * D_;
  float s = 1e-6f;
#pragma unroll
  for (int d = 0; d < D_; ++d) s += bf2f(qrow[d]) * ks[d];
  denom[idx] = s;
}

// ---------------------------------------------------------------------------
// 7) out_num = Q' * kv, normalized by denom; writes att bf16 (B*L, C) layout.
//    grid = (L/256, B*H); wave tile 32(l) x 64(e); K = D = 64 (2 WMMA steps).
// ---------------------------------------------------------------------------
__global__ __launch_bounds__(256) void attn_out_kernel(
    const unsigned short* __restrict__ qp, const unsigned short* __restrict__ kvT,
    const float* __restrict__ denom, unsigned short* __restrict__ att) {
  const int bh   = blockIdx.y;
  const int b    = bh >> 4, h = bh & 15;
  const int lane = threadIdx.x & 31;
  const int wave = threadIdx.x >> 5;
  const int m0   = blockIdx.x * 256 + wave * 32;  // l offset
  const int lr   = lane & 15;
  const int khalf = (lane >> 4) * 8;

  v8f acc[2][4];
#pragma unroll
  for (int i = 0; i < 2; ++i)
#pragma unroll
    for (int j = 0; j < 4; ++j) acc[i][j] = {};

#pragma unroll
  for (int k0 = 0; k0 < D_; k0 += 32) {
    FragBf aF[2], bF[4];
#pragma unroll
    for (int i = 0; i < 2; ++i) {
      const unsigned short* p =
          qp + ((size_t)(b * L_ + m0 + i * 16 + lr)) * C_ + h * D_ + k0 + khalf;
      aF[i].u[0] = *(const ushort8*)(p);
      aF[i].u[1] = *(const ushort8*)(p + 16);
    }
#pragma unroll
    for (int j = 0; j < 4; ++j) {
      const unsigned short* p =
          kvT + ((size_t)bh * D_ + j * 16 + lr) * D_ + k0 + khalf;
      bF[j].u[0] = *(const ushort8*)(p);
      bF[j].u[1] = *(const ushort8*)(p + 16);
    }
#pragma unroll
    for (int i = 0; i < 2; ++i)
#pragma unroll
      for (int j = 0; j < 4; ++j)
        acc[i][j] = __builtin_amdgcn_wmma_f32_16x16x32_bf16(
            false, aF[i].bf, false, bF[j].bf, (short)0, acc[i][j], false, false);
  }

  const int rbase = (lane >> 4) * 8;
#pragma unroll
  for (int i = 0; i < 2; ++i)
#pragma unroll
    for (int r = 0; r < 8; ++r) {
      const int l   = m0 + i * 16 + rbase + r;
      const float dn = denom[(size_t)bh * L_ + l];
      const float inv = 1.f / dn;
#pragma unroll
      for (int j = 0; j < 4; ++j) {
        const int e = j * 16 + lr;
        att[((size_t)(b * L_ + l)) * C_ + h * D_ + e] = f2bf(acc[i][j][r] * inv);
      }
    }
}

// ---------------------------------------------------------------------------
// launch
// ---------------------------------------------------------------------------
extern "C" void kernel_launch(void* const* d_in, const int* in_sizes, int n_in,
                              void* d_out, int out_size, void* d_ws, size_t ws_size,
                              hipStream_t stream) {
  const float* x  = (const float*)d_in[0];
  const float* Wq = (const float*)d_in[1];
  const float* bq = (const float*)d_in[2];
  const float* Wk = (const float*)d_in[3];
  const float* bk = (const float*)d_in[4];
  const float* Wv = (const float*)d_in[5];
  const float* bv = (const float*)d_in[6];
  const float* Wo = (const float*)d_in[7];
  const float* bo = (const float*)d_in[8];
  float* out = (float*)d_out;

  const size_t NE = (size_t)B_ * L_ * C_;  // 16,777,216
  char* ws = (char*)d_ws;
  size_t off = 0;
  auto carve = [&](size_t bytes) -> char* {
    char* p = ws + off;
    off += (bytes + 255) & ~(size_t)255;
    return p;
  };

  unsigned short* xb   = (unsigned short*)carve(NE * 2);
  unsigned short* wqT  = (unsigned short*)carve((size_t)C_ * C_ * 2);
  unsigned short* wkT  = (unsigned short*)carve((size_t)C_ * C_ * 2);
  unsigned short* wvT  = (unsigned short*)carve((size_t)C_ * C_ * 2);
  unsigned short* woT  = (unsigned short*)carve((size_t)C_ * C_ * 2);
  float*          qf   = (float*)carve(NE * 4);
  float*          kf   = (float*)carve(NE * 4);
  float*          vf   = (float*)carve(NE * 4);
  unsigned short* qp   = (unsigned short*)carve(NE * 2);
  unsigned short* kpT  = (unsigned short*)carve(NE * 2);
  unsigned short* vT   = (unsigned short*)carve(NE * 2);
  unsigned short* attb = (unsigned short*)carve(NE * 2);
  unsigned short* kvT  = (unsigned short*)carve((size_t)B_ * H_ * D_ * D_ * 2);
  float*          ksum = (float*)carve((size_t)B_ * H_ * D_ * 4);
  float*          dnm  = (float*)carve((size_t)B_ * H_ * L_ * 4);

  const dim3 b256(256);

  cvt_bf16_kernel<<<(unsigned)((NE + 255) / 256), b256, 0, stream>>>(x, xb, NE);

  const dim3 tb(32, 8), tg(C_ / 32, C_ / 32);
  transpose_w_kernel<<<tg, tb, 0, stream>>>(Wq, wqT);
  transpose_w_kernel<<<tg, tb, 0, stream>>>(Wk, wkT);
  transpose_w_kernel<<<tg, tb, 0, stream>>>(Wv, wvT);
  transpose_w_kernel<<<tg, tb, 0, stream>>>(Wo, woT);

  const dim3 gg(M_ / 256, C_ / 64);  // (64, 16)
  gemm_bf16_tn_kernel<<<gg, b256, 0, stream>>>(xb, wqT, bq, qf, M_, C_, C_);
  gemm_bf16_tn_kernel<<<gg, b256, 0, stream>>>(xb, wkT, bk, kf, M_, C_, C_);
  gemm_bf16_tn_kernel<<<gg, b256, 0, stream>>>(xb, wvT, bv, vf, M_, C_, C_);

  softmax_pack_kernel<<<(B_ * L_ * D_) / 256, b256, 0, stream>>>(qf, kf, vf, qp,
                                                                 kpT, vT);
  kv_state_kernel<<<B_ * H_, b256, 0, stream>>>(kpT, vT, kvT, ksum);
  denom_kernel<<<(B_ * H_ * L_) / 256, b256, 0, stream>>>(qp, ksum, dnm);
  attn_out_kernel<<<dim3(L_ / 256, B_ * H_), b256, 0, stream>>>(qp, kvT, dnm, attb);

  gemm_bf16_tn_kernel<<<gg, b256, 0, stream>>>(attb, woT, bo, out, M_, C_, C_);
}
